// RelationalProposalModule_63608465653823
// MI455X (gfx1250) — matrix operand order repaired and checked by reading, not compiled, not run on hardware
//
#include <hip/hip_runtime.h>
#include <hip/hip_bf16.h>
#include <math.h>

// ---------------------------------------------------------------------------
// Problem constants (match reference)
// ---------------------------------------------------------------------------
#define BB   32
#define CC   128
#define NN   1024
#define DKK  64
#define NRR  4
#define NSCC 10
#define OUTC 79
#define M4   (NRR * NN)          // 4096
#define BNP  (BB * NN)           // 32768 points
#define EPSV 1e-5f

typedef __attribute__((ext_vector_type(16))) _Float16 v16h;
typedef __attribute__((ext_vector_type(8)))  float    v8f;
typedef __attribute__((ext_vector_type(8)))  _Float16 h8;

// ---------------------------------------------------------------------------
// CDNA5 async global->LDS copy (ASYNCcnt path). The builtin takes
// (v4i32 AS1* src, v4i32 AS3* dst, imm offset, imm cpol).
// ---------------------------------------------------------------------------
typedef int i4v __attribute__((vector_size(16)));
typedef __attribute__((address_space(1))) i4v gvi4;
typedef __attribute__((address_space(3))) i4v lvi4;

__device__ __forceinline__ gvi4* to_glb(const void* p)
{
    return (gvi4*)(unsigned long long)(uintptr_t)p;
}
__device__ __forceinline__ lvi4* to_lds(void* p)
{
    return (lvi4*)(unsigned int)(uintptr_t)p;   // generic->LDS: low 32 bits
}

#if __has_builtin(__builtin_amdgcn_global_load_async_to_lds_b128)
#define HAVE_ASYNC_LDS 1
#define ASYNC_CP_B128(g, l) __builtin_amdgcn_global_load_async_to_lds_b128((g), (l), 0, 0)
#else
#define HAVE_ASYNC_LDS 0
#endif

#if __has_builtin(__builtin_amdgcn_s_wait_asynccnt)
#define WAIT_ASYNC0() __builtin_amdgcn_s_wait_asynccnt(0)
#else
#define WAIT_ASYNC0() asm volatile("s_wait_asynccnt 0x0" ::: "memory")
#endif

// ---------------------------------------------------------------------------
// f16 GEMM: C[z] = A[z] * BT[z]^T, A [M x K] rm f16, BT [N x K] rm f16
// (B stored n-major). 4 waves/block, 64x64 block tile, 4 WMMA acc / wave.
// K mult of 32, N mult of 64; GUARD_M=false requires M mult of 64.
// Double-buffered LDS pipeline; async-to-LDS staging when available.
// ---------------------------------------------------------------------------
template <bool GUARD_M>
__launch_bounds__(128)
__global__ void gemm_f16(const _Float16* __restrict__ A, int lda, long Asb,
                         const _Float16* __restrict__ BT, int ldb, long Bsb,
                         float* __restrict__ C32, int ldc, long Csb,
                         _Float16* __restrict__ C16, int ldc16, long C16sb, int c16t,
                         int M, int N, int K,
                         const float* __restrict__ alpha,
                         const float* __restrict__ beta,
                         int relu)
{
    __shared__ _Float16 Ash[2][64][32];
    __shared__ _Float16 Bsh[2][64][32];       // BT tile: [n][k]

    const int tid  = threadIdx.x;             // 0..127
    const int lane = tid & 31;
    const int w    = tid >> 5;                // wave 0..3
    const int m0   = blockIdx.y * 64;
    const int n0   = blockIdx.x * 64;
    const _Float16* Ab = A  + (long)blockIdx.z * Asb;
    const _Float16* Bb = BT + (long)blockIdx.z * Bsb;

    // staging geometry: thread owns rows r0 and r0+32, k-chunk kc (8 halves)
    const int r0 = tid >> 2;
    const int kc = (tid & 3) * 8;

    v8f acc[4] = {};

    const int fr = lane & 15;
    const int hi = lane >> 4;

    // fragment compute on buffer p
    auto compute = [&](int p) {
        h8 alo = *(const h8*)&Ash[p][w * 16 + fr][hi * 8];
        h8 ahi = *(const h8*)&Ash[p][w * 16 + fr][16 + hi * 8];
        v16h af;
        #pragma unroll
        for (int t = 0; t < 8; ++t) { af[t] = alo[t]; af[8 + t] = ahi[t]; }
        #pragma unroll
        for (int nt = 0; nt < 4; ++nt) {
            h8 blo = *(const h8*)&Bsh[p][nt * 16 + fr][hi * 8];
            h8 bhi = *(const h8*)&Bsh[p][nt * 16 + fr][16 + hi * 8];
            v16h bf;
            #pragma unroll
            for (int t = 0; t < 8; ++t) { bf[t] = blo[t]; bf[8 + t] = bhi[t]; }
            acc[nt] = __builtin_amdgcn_wmma_f32_16x16x32_f16(
                          false, af, false, bf, (short)0, acc[nt], false, false);
        }
    };

#if HAVE_ASYNC_LDS
    if constexpr (!GUARD_M) {
        // ---- async ping-pong pipeline (ASYNCcnt) ----
        auto stage = [&](int p, int k0) {
            ASYNC_CP_B128(to_glb(Ab + (long)(m0 + r0)      * lda + k0 + kc), to_lds(&Ash[p][r0][kc]));
            ASYNC_CP_B128(to_glb(Ab + (long)(m0 + r0 + 32) * lda + k0 + kc), to_lds(&Ash[p][r0 + 32][kc]));
            ASYNC_CP_B128(to_glb(Bb + (long)(n0 + r0)      * ldb + k0 + kc), to_lds(&Bsh[p][r0][kc]));
            ASYNC_CP_B128(to_glb(Bb + (long)(n0 + r0 + 32) * ldb + k0 + kc), to_lds(&Bsh[p][r0 + 32][kc]));
        };
        stage(0, 0);
        WAIT_ASYNC0();
        __syncthreads();
        int p = 0;
        for (int k0 = 0; k0 < K; k0 += 32) {
            const bool hasNext = (k0 + 32) < K;
            if (hasNext) stage(p ^ 1, k0 + 32);   // copy engine runs under the WMMAs
            compute(p);
            if (hasNext) WAIT_ASYNC0();
            __syncthreads();
            p ^= 1;
        }
    } else
#endif
    {
        // ---- synchronous pipeline, register double buffer ----
        h8 ra0, ra1, rb0, rb1;
        auto loadRegs = [&](int k0) {
            if (GUARD_M) {
                if (m0 + r0 < M)      ra0 = *(const h8*)(Ab + (long)(m0 + r0) * lda + k0 + kc);
                else                  ra0 = h8{};
                if (m0 + r0 + 32 < M) ra1 = *(const h8*)(Ab + (long)(m0 + r0 + 32) * lda + k0 + kc);
                else                  ra1 = h8{};
            } else {
                ra0 = *(const h8*)(Ab + (long)(m0 + r0) * lda + k0 + kc);
                ra1 = *(const h8*)(Ab + (long)(m0 + r0 + 32) * lda + k0 + kc);
            }
            rb0 = *(const h8*)(Bb + (long)(n0 + r0) * ldb + k0 + kc);
            rb1 = *(const h8*)(Bb + (long)(n0 + r0 + 32) * ldb + k0 + kc);
        };
        auto storeRegs = [&](int p) {
            *(h8*)&Ash[p][r0][kc]      = ra0;
            *(h8*)&Ash[p][r0 + 32][kc] = ra1;
            *(h8*)&Bsh[p][r0][kc]      = rb0;
            *(h8*)&Bsh[p][r0 + 32][kc] = rb1;
        };
        loadRegs(0);
        storeRegs(0);
        __syncthreads();
        int p = 0;
        for (int k0 = 0; k0 < K; k0 += 32) {
            const bool hasNext = (k0 + 32) < K;
            if (hasNext) loadRegs(k0 + 32);       // global loads in flight during WMMAs
            compute(p);
            if (hasNext) storeRegs(p ^ 1);
            __syncthreads();
            p ^= 1;
        }
    }

    // --- epilogue; C/D layout: vgpr j, lane L -> m = j + 8*(L>=16), n = L&15 ---
    const int col = lane & 15;
    const int mb  = m0 + w * 16 + ((lane >> 4) << 3);
    #pragma unroll
    for (int nt = 0; nt < 4; ++nt) {
        const int nc = n0 + nt * 16 + col;
        float vv[8];
        #pragma unroll
        for (int j = 0; j < 8; ++j) {
            int m = mb + j;
            float v = acc[nt][j];
            if (alpha) v *= alpha[GUARD_M ? (m < M ? m : 0) : m];
            if (beta)  v += beta [GUARD_M ? (m < M ? m : 0) : m];
            if (relu && v < 0.0f) v = 0.0f;
            vv[j] = v;
        }
        if (C32) {
            float* Cb = C32 + (long)blockIdx.z * Csb;
            #pragma unroll
            for (int j = 0; j < 8; ++j) {
                int m = mb + j;
                if (!GUARD_M || m < M) Cb[(long)m * ldc + nc] = vv[j];
            }
        }
        if (C16) {
            _Float16* Cb = C16 + (long)blockIdx.z * C16sb;
            if (c16t) {                       // [n][m] row-major, 8 contiguous m
                h8 hv;
                #pragma unroll
                for (int j = 0; j < 8; ++j) hv[j] = (_Float16)vv[j];
                *(h8*)(Cb + (long)nc * ldc16 + mb) = hv;
            } else {                          // [m][n] row-major
                #pragma unroll
                for (int j = 0; j < 8; ++j) {
                    int m = mb + j;
                    if (!GUARD_M || m < M) Cb[(long)m * ldc16 + nc] = (_Float16)vv[j];
                }
            }
        }
    }
}

// ---------------------------------------------------------------------------
// Fold eval-mode BN (+optional preceding conv bias) into per-channel affine
// ---------------------------------------------------------------------------
__global__ void bn_fold(const float* __restrict__ g, const float* __restrict__ b,
                        const float* __restrict__ m, const float* __restrict__ v,
                        const float* __restrict__ bias,
                        float* __restrict__ scale, float* __restrict__ shift, int n)
{
    int i = blockIdx.x * blockDim.x + threadIdx.x;
    if (i >= n) return;
    float s = g[i] * rsqrtf(v[i] + EPSV);
    float sh = b[i] - m[i] * s;
    if (bias) sh += s * bias[i];
    scale[i] = s;
    shift[i] = sh;
}

// ---------------------------------------------------------------------------
// f32 -> f16 copy (weights)
// ---------------------------------------------------------------------------
__global__ void to_f16(const float* __restrict__ src, _Float16* __restrict__ dst, int n)
{
    int i = blockIdx.x * 256 + threadIdx.x;
    if (i < n) dst[i] = (_Float16)src[i];
}

// ---------------------------------------------------------------------------
// geo = [center(3), mean[cls] + resn[cls]*mean[cls]], cls = argmax scores
// ---------------------------------------------------------------------------
__global__ void geo_kernel(const float* __restrict__ center,
                           const float* __restrict__ scores,
                           const float* __restrict__ resn,
                           const float* __restrict__ meansz,
                           float* __restrict__ geo)
{
    int i = blockIdx.x * blockDim.x + threadIdx.x;   // b*N + n
    if (i >= BNP) return;
    const float* sc = scores + (long)i * NSCC;
    int cls = 0; float best = sc[0];
    #pragma unroll
    for (int k = 1; k < NSCC; ++k) { float s = sc[k]; if (s > best) { best = s; cls = k; } }
    float* g = geo + (long)i * 6;
    const float* ctr = center + (long)i * 3;
    g[0] = ctr[0]; g[1] = ctr[1]; g[2] = ctr[2];
    const float* rr = resn + ((long)i * NSCC + cls) * 3;
    #pragma unroll
    for (int j = 0; j < 3; ++j) {
        float ms = meansz[cls * 3 + j];
        g[3 + j] = ms + rr[j] * ms;
    }
}

// ---------------------------------------------------------------------------
// Position-embedding MLP fused with f = proposal + pe. One block(128) / point.
// Writes f (f32, [B,C,N]) and fT16 (f16, [B,N,C]) for the QKV GEMM A-operand.
// ---------------------------------------------------------------------------
__global__ void pe_kernel(const float* __restrict__ geo,
                          const float* __restrict__ w1, const float* __restrict__ w2,
                          const float* __restrict__ b2,
                          const float* __restrict__ pscale, const float* __restrict__ pshift,
                          const float* __restrict__ pfin,
                          float* __restrict__ f, _Float16* __restrict__ fT16)
{
    int bn = blockIdx.x;                 // b*N + n
    int b  = bn >> 10;
    int n  = bn & (NN - 1);
    int t  = threadIdx.x;                // channel 0..127
    __shared__ float gsh[6];
    __shared__ float hsh[CC];
    if (t < 6) gsh[t] = geo[(long)bn * 6 + t];
    __syncthreads();
    float h = 0.0f;
    #pragma unroll
    for (int i = 0; i < 6; ++i) h += w1[t * 6 + i] * gsh[i];
    h = fmaxf(h * pscale[t] + pshift[t], 0.0f);
    hsh[t] = h;
    __syncthreads();
    float pe = b2[t];
    #pragma unroll 8
    for (int j = 0; j < CC; ++j) pe += w2[t * CC + j] * hsh[j];
    float fv = pfin[(long)b * (CC * NN) + (long)t * NN + n] + pe;
    f[(long)b * (CC * NN) + (long)t * NN + n] = fv;
    fT16[((long)b * NN + n) * CC + t] = (_Float16)fv;
}

// ---------------------------------------------------------------------------
// Row softmax: read S (f32), write P (f16, [n][m] row-major). 256 thr, 4/thr.
// ---------------------------------------------------------------------------
__global__ void softmax_p16(const float* __restrict__ S, _Float16* __restrict__ P)
{
    int row = blockIdx.x;
    const float* r = S + (long)row * NN;
    _Float16*    p = P + (long)row * NN;
    int t = threadIdx.x;
    float v[4];
    #pragma unroll
    for (int i = 0; i < 4; ++i) v[i] = r[t + i * 256];
    __shared__ float red[256];
    float mx = fmaxf(fmaxf(v[0], v[1]), fmaxf(v[2], v[3]));
    red[t] = mx; __syncthreads();
    for (int s = 128; s > 0; s >>= 1) { if (t < s) red[t] = fmaxf(red[t], red[t + s]); __syncthreads(); }
    mx = red[0]; __syncthreads();
    float sum = 0.0f;
    #pragma unroll
    for (int i = 0; i < 4; ++i) { v[i] = expf(v[i] - mx); sum += v[i]; }
    red[t] = sum; __syncthreads();
    for (int s = 128; s > 0; s >>= 1) { if (t < s) red[t] += red[t + s]; __syncthreads(); }
    float inv = 1.0f / red[0];
    #pragma unroll
    for (int i = 0; i < 4; ++i) p[t + i * 256] = (_Float16)(v[i] * inv);
}

// ---------------------------------------------------------------------------
// x = rawreshape(O[B,N,C]->[B,C,N]) + f; LayerNorm over channels per (b,n);
// write cat16 in n-major f16 layout: cat16[b][r*NN+n][c].
// ---------------------------------------------------------------------------
__global__ void add_ln(const float* __restrict__ O, const float* __restrict__ f,
                       const float* __restrict__ lng, const float* __restrict__ lnb,
                       _Float16* __restrict__ cat16, int r)
{
    int bn = blockIdx.x;
    int b  = bn >> 10;
    int n  = bn & (NN - 1);
    int c  = threadIdx.x;                // 0..127
    long idx = (long)b * (CC * NN) + (long)c * NN + n;  // raw-reshape index
    float x = O[idx] + f[idx];
    __shared__ float red[CC];
    red[c] = x; __syncthreads();
    for (int s = 64; s > 0; s >>= 1) { if (c < s) red[c] += red[c + s]; __syncthreads(); }
    float mu = red[0] * (1.0f / CC);
    __syncthreads();
    float d = x - mu;
    red[c] = d * d; __syncthreads();
    for (int s = 64; s > 0; s >>= 1) { if (c < s) red[c] += red[c + s]; __syncthreads(); }
    float var = red[0] * (1.0f / CC);
    float y = d * rsqrtf(var + EPSV) * lng[c] + lnb[c];
    cat16[((long)b * M4 + (long)r * NN + n) * CC + c] = (_Float16)y;
}

// ---------------------------------------------------------------------------
// Host-side launcher
// ---------------------------------------------------------------------------
extern "C" void kernel_launch(void* const* d_in, const int* in_sizes, int n_in,
                              void* d_out, int out_size, void* d_ws, size_t ws_size,
                              hipStream_t stream)
{
    (void)in_sizes; (void)n_in; (void)out_size; (void)ws_size;

    // ---- inputs (setup_inputs order; params pytree flattened, keys sorted) ----
    const float* proposal = (const float*)d_in[0];      // [B,C,N]
    const float* center   = (const float*)d_in[1];      // [B,N,3]
    const float* scores   = (const float*)d_in[2];      // [B,N,10]
    const float* resn     = (const float*)d_in[3];      // [B,N,10,3]
    const float* meansz   = (const float*)d_in[4];      // [10,3]
    const float* ft_b1  = (const float*)d_in[5];
    const float* ft_b2  = (const float*)d_in[6];
    const float* ft_bnb = (const float*)d_in[7];
    const float* ft_bng = (const float*)d_in[8];
    const float* ft_bnm = (const float*)d_in[9];
    const float* ft_bnv = (const float*)d_in[10];
    const float* ft_w1  = (const float*)d_in[11];
    const float* ft_w2  = (const float*)d_in[12];
    const float* pg_b1   = (const float*)d_in[13];
    const float* pg_b2   = (const float*)d_in[14];
    const float* pg_b3   = (const float*)d_in[15];
    const float* pg_bn1b = (const float*)d_in[16];
    const float* pg_bn1g = (const float*)d_in[17];
    const float* pg_bn1m = (const float*)d_in[18];
    const float* pg_bn1v = (const float*)d_in[19];
    const float* pg_bn2b = (const float*)d_in[20];
    const float* pg_bn2g = (const float*)d_in[21];
    const float* pg_bn2m = (const float*)d_in[22];
    const float* pg_bn2v = (const float*)d_in[23];
    const float* pg_w1   = (const float*)d_in[24];
    const float* pg_w2   = (const float*)d_in[25];
    const float* pg_w3   = (const float*)d_in[26];
    const float* pos_b1  = (const float*)d_in[27];
    const float* pos_b2  = (const float*)d_in[28];
    const float* pos_bnb = (const float*)d_in[29];
    const float* pos_bng = (const float*)d_in[30];
    const float* pos_bnm = (const float*)d_in[31];
    const float* pos_bnv = (const float*)d_in[32];
    const float* pos_w1  = (const float*)d_in[33];
    const float* pos_w2  = (const float*)d_in[34];
    const float* rK[NRR], *rQ[NRR], *rV[NRR], *rLnB[NRR], *rLnG[NRR];
    for (int r = 0; r < NRR; ++r) {
        int base = 35 + r * 5;
        rK[r]   = (const float*)d_in[base + 0];
        rQ[r]   = (const float*)d_in[base + 1];
        rV[r]   = (const float*)d_in[base + 2];
        rLnB[r] = (const float*)d_in[base + 3];
        rLnG[r] = (const float*)d_in[base + 4];
    }

    // ---- workspace layout ----
    float* wsf = (float*)d_ws;
    float* geo = wsf;                         //   196,608 f32
    float* f   = wsf + 196608;                // 4,194,304 f32 [B,C,N]
    float* S32 = wsf + 4390912;               // 1,048,576 f32 [N,N] (per r,b)
    float* O32 = wsf + 5439488;               // 4,194,304 f32 [B,N,C]
    float* eff = wsf + 9633792;               //     1,024 f32
    float* posS = eff + 0,   *posH = eff + 128;
    float* ftS  = eff + 256, *ftH  = eff + 384;
    float* pg1S = eff + 512, *pg1H = eff + 640;
    float* pg2S = eff + 768, *pg2H = eff + 896;

    _Float16* wsh  = (_Float16*)(wsf + 9634816);
    _Float16* fT16 = wsh;                     //  4,194,304 h [B,N,C]
    _Float16* Q16  = wsh + 4194304;           //  2,097,152 h [B,N,DK]
    _Float16* K16  = wsh + 6291456;           //  2,097,152 h [B,N,DK]
    _Float16* V16t = wsh + 8388608;           //  4,194,304 h [B,C,N] (n-major for O)
    _Float16* P16  = wsh + 12582912;          //  1,048,576 h [N,N]
    _Float16* cat16= wsh + 13631488;          // 16,777,216 h [B,M4,C] (n-major)
    _Float16* H116 = wsh + 30408704;          // 16,777,216 h [B,M4,C] (n-major)
    _Float16* pf16 = wsh + 47185920;          // 16,777,216 h [B,M4,C] (n-major)
    _Float16* relw = wsh + 63963136;          // 4 * (8192+8192+16384)
    _Float16* ftw1_16 = wsh + 64094208;       // 16384
    _Float16* ftw2_16 = ftw1_16 + 16384;
    _Float16* pgw1_16 = ftw2_16 + 16384;
    _Float16* pgw2_16 = pgw1_16 + 16384;
    _Float16* pgw3_16 = pgw2_16 + 16384;      // 10112

    float* net_out = (float*)d_out;                        // [B,79,4096]
    float* pf_out  = (float*)d_out + (long)BB * OUTC * M4; // [B,128,4096]

    const long CN  = (long)CC * NN;   // 131072
    const long NDK = (long)NN * DKK;  // 65536
    const long CM4 = (long)CC * M4;   // 524288

    // ---- BN folds ----
    bn_fold<<<1, 128, 0, stream>>>(pos_bng, pos_bnb, pos_bnm, pos_bnv, pos_b1, posS, posH, CC);
    bn_fold<<<1, 128, 0, stream>>>(ft_bng,  ft_bnb,  ft_bnm,  ft_bnv,  ft_b1,  ftS,  ftH,  CC);
    bn_fold<<<1, 128, 0, stream>>>(pg_bn1g, pg_bn1b, pg_bn1m, pg_bn1v, pg_b1,  pg1S, pg1H, CC);
    bn_fold<<<1, 128, 0, stream>>>(pg_bn2g, pg_bn2b, pg_bn2m, pg_bn2v, pg_b2,  pg2S, pg2H, CC);

    // ---- weight f16 conversions (raw layouts are already the n-major forms) ----
    for (int r = 0; r < NRR; ++r) {
        _Float16* base = relw + (long)r * 32768;
        to_f16<<<32, 256, 0, stream>>>(rQ[r], base,          8192);
        to_f16<<<32, 256, 0, stream>>>(rK[r], base + 8192,   8192);
        to_f16<<<64, 256, 0, stream>>>(rV[r], base + 16384, 16384);
    }
    to_f16<<<64, 256, 0, stream>>>(ft_w1, ftw1_16, 16384);
    to_f16<<<64, 256, 0, stream>>>(ft_w2, ftw2_16, 16384);
    to_f16<<<64, 256, 0, stream>>>(pg_w1, pgw1_16, 16384);
    to_f16<<<64, 256, 0, stream>>>(pg_w2, pgw2_16, 16384);
    to_f16<<<40, 256, 0, stream>>>(pg_w3, pgw3_16, OUTC * CC);

    // ---- geo + position embedding + f ----
    geo_kernel<<<(BNP + 255) / 256, 256, 0, stream>>>(center, scores, resn, meansz, geo);
    pe_kernel<<<BNP, 128, 0, stream>>>(geo, pos_w1, pos_w2, pos_b2, posS, posH, proposal, f, fT16);

    // ---- relation units ----
    for (int r = 0; r < NRR; ++r) {
        const _Float16* wq16 = relw + (long)r * 32768;
        const _Float16* wk16 = wq16 + 8192;
        const _Float16* wv16 = wq16 + 16384;

        // Q[b,n,d] (std layout -> A of S); K[b,m,d] (std layout == BT of S)
        gemm_f16<false><<<dim3(1, 16, BB), 128, 0, stream>>>(
            fT16, CC, CN,  wq16, CC, 0L,
            nullptr, 0, 0L,  Q16, DKK, NDK, 0,
            NN, DKK, CC, nullptr, nullptr, 0);
        gemm_f16<false><<<dim3(1, 16, BB), 128, 0, stream>>>(
            fT16, CC, CN,  wk16, CC, 0L,
            nullptr, 0, 0L,  K16, DKK, NDK, 0,
            NN, DKK, CC, nullptr, nullptr, 0);
        // V stored n-major [c][m] (BT of O)
        gemm_f16<false><<<dim3(2, 16, BB), 128, 0, stream>>>(
            fT16, CC, CN,  wv16, CC, 0L,
            nullptr, 0, 0L,  V16t, NN, CN, 1,
            NN, CC, CC, nullptr, nullptr, 0);

        for (int b = 0; b < BB; ++b) {
            // S[n,m] = sum_d Q[n,d] K[m,d]
            gemm_f16<false><<<dim3(16, 16, 1), 128, 0, stream>>>(
                Q16 + (long)b * NDK, DKK, 0L,  K16 + (long)b * NDK, DKK, 0L,
                S32, NN, 0L,  nullptr, 0, 0L, 0,
                NN, NN, DKK, nullptr, nullptr, 0);
            softmax_p16<<<NN, 256, 0, stream>>>(S32, P16);
            // O[n,c] = sum_m P[n,m] V[m,c]
            gemm_f16<false><<<dim3(2, 16, 1), 128, 0, stream>>>(
                P16, NN, 0L,  V16t + (long)b * CN, NN, 0L,
                O32 + (long)b * CN, CC, 0L,  nullptr, 0, 0L, 0,
                NN, CC, NN, nullptr, nullptr, 0);
        }
        add_ln<<<BNP, CC, 0, stream>>>(O32, f, rLnG[r], rLnB[r], cat16, r);
    }

    // ---- ft2 ----
    gemm_f16<false><<<dim3(64, 2, BB), 128, 0, stream>>>(
        ftw1_16, CC, 0L,  cat16, CC, CM4,
        nullptr, 0, 0L,  H116, CC, CM4, 1,
        CC, M4, CC, ftS, ftH, 1);
    gemm_f16<false><<<dim3(64, 2, BB), 128, 0, stream>>>(
        ftw2_16, CC, 0L,  H116, CC, CM4,
        pf_out, M4, CM4,  pf16, CC, CM4, 1,
        CC, M4, CC, nullptr, ft_b2, 0);

    // ---- pg ----
    gemm_f16<false><<<dim3(64, 2, BB), 128, 0, stream>>>(
        pgw1_16, CC, 0L,  pf16, CC, CM4,
        nullptr, 0, 0L,  H116, CC, CM4, 1,
        CC, M4, CC, pg1S, pg1H, 1);
    gemm_f16<false><<<dim3(64, 2, BB), 128, 0, stream>>>(
        pgw2_16, CC, 0L,  H116, CC, CM4,
        nullptr, 0, 0L,  cat16, CC, CM4, 1,
        CC, M4, CC, pg2S, pg2H, 1);
    gemm_f16<true><<<dim3(64, 2, BB), 128, 0, stream>>>(
        pgw3_16, CC, 0L,  cat16, CC, CM4,
        net_out, M4, (long)OUTC * M4,  nullptr, 0, 0L, 0,
        OUTC, M4, CC, nullptr, pg_b3, 0);
}
